// HyenaPositioning_75033078661223
// MI455X (gfx1250) — matrix operand order repaired and checked by reading, not compiled, not run on hardware
//
#include <hip/hip_runtime.h>
#include <cstdint>
#include <cstddef>

// ---------------------------------------------------------------------------
// Problem constants (from reference)
// ---------------------------------------------------------------------------
#define BB    8
#define LL    4096
#define INW   3
#define HB    512
#define DEPTH 8
#define FHH   64
#define H3    1536
#define OUTD  2
#define MROWS (BB * LL)          // 32768 token rows
#define HHALF 256                // H/2

typedef __attribute__((ext_vector_type(16))) __bf16 v16bf;
typedef __attribute__((ext_vector_type(8)))  float  v8f;
typedef __attribute__((ext_vector_type(4)))  float  f32x4;

struct cplx { float x, y; };

__device__ __forceinline__ float gelu_exact(float x) {
    return 0.5f * x * (1.0f + erff(x * 0.70710678118654752f));
}

// ---------------------------------------------------------------------------
// Embed: h = x @ in_w + in_b + sinusoidal PE + edge embedding
// ---------------------------------------------------------------------------
__global__ __launch_bounds__(256) void k_embed(
    const float* __restrict__ x, const int* __restrict__ edge_ids,
    const float* __restrict__ in_w, const float* __restrict__ in_b,
    const float* __restrict__ edge_emb, float* __restrict__ h)
{
    int idx = blockIdx.x * 256 + threadIdx.x;            // over MROWS*HB
    if (idx >= MROWS * HB) return;
    int c   = idx & (HB - 1);
    int row = idx >> 9;                                  // b*L + l
    int l   = row & (LL - 1);
    int b   = row >> 12;
    const float* xr = x + (size_t)row * INW;
    float acc = in_b[c]
              + xr[0] * in_w[0 * HB + c]
              + xr[1] * in_w[1 * HB + c]
              + xr[2] * in_w[2 * HB + c];
    int i2 = c >> 1;                                     // interleaved sin/cos
    float div = __expf((float)(2 * i2) * (-9.2103403719761836f / (float)HB));
    float ang = (float)l * div;
    acc += (c & 1) ? __cosf(ang) : __sinf(ang);
    acc += edge_emb[edge_ids[b] * HB + c];
    h[idx] = acc;
}

// ---------------------------------------------------------------------------
// LayerNorm over H=512 -> bf16 output (one block per row, 2 elems/thread)
// ---------------------------------------------------------------------------
__global__ __launch_bounds__(256) void k_ln(
    const float* __restrict__ x, const float* __restrict__ g,
    const float* __restrict__ b, __bf16* __restrict__ o)
{
    __shared__ float red[256];
    const int row = blockIdx.x;
    const int t = threadIdx.x;
    const float* xr = x + (size_t)row * HB;
    float v0 = xr[t], v1 = xr[t + 256];
    red[t] = v0 + v1;
    __syncthreads();
    for (int s = 128; s > 0; s >>= 1) { if (t < s) red[t] += red[t + s]; __syncthreads(); }
    float mean = red[0] * (1.0f / HB);
    __syncthreads();
    float d0 = v0 - mean, d1 = v1 - mean;
    red[t] = d0 * d0 + d1 * d1;
    __syncthreads();
    for (int s = 128; s > 0; s >>= 1) { if (t < s) red[t] += red[t + s]; __syncthreads(); }
    float inv = rsqrtf(red[0] * (1.0f / HB) + 1e-5f);
    __bf16* orow = o + (size_t)row * HB;
    orow[t]       = (__bf16)(d0 * inv * g[t]       + b[t]);
    orow[t + 256] = (__bf16)(d1 * inv * g[t + 256] + b[t + 256]);
}

// ---------------------------------------------------------------------------
// Pack fp32 weight (K x N row-major) into WMMA B-fragment tiles (bf16).
// Fragment layout used here (contiguous k, A matches): lanes 0-15 hold K=k0..k0+15, lanes 16-31
// hold K=k0+16..k0+31, column = lane&15. Each lane's 16 values contiguous.
// ---------------------------------------------------------------------------
__global__ __launch_bounds__(256) void k_pack(
    const float* __restrict__ W, __bf16* __restrict__ Bp, int K, int N)
{
    int idx = blockIdx.x * 256 + threadIdx.x;   // over (K/32)*(N/16)*32
    int total = (K >> 5) * (N >> 4) * 32;
    if (idx >= total) return;
    int lane  = idx & 31;
    int tile  = idx >> 5;
    int nT    = N >> 4;
    int ntile = tile % nT;
    int kt    = tile / nT;
    int n     = (ntile << 4) + (lane & 15);
    int kbase = (kt << 5) + ((lane >> 4) << 4);
    __bf16* dst = Bp + ((size_t)idx << 4);
    #pragma unroll
    for (int j = 0; j < 16; ++j)
        dst[j] = (__bf16)W[(size_t)(kbase + j) * N + n];
}

// ---------------------------------------------------------------------------
// bf16 WMMA GEMM: C[M,N] = A[M,K] @ B + bias (+addsrc) (+gelu)
// Block = (32,8) = 8 waves; each wave: 16 rows x 64 cols.
// 2-stage software pipeline over the K loop (K/32 iterations, must be even):
// while WMMA consumes stage s, the b128 loads for stage s^1 are in flight.
// ---------------------------------------------------------------------------
__global__ __launch_bounds__(256) void k_gemm(
    const __bf16* __restrict__ A, const __bf16* __restrict__ Bp,
    const float* __restrict__ bias, const float* __restrict__ addsrc,
    float* __restrict__ C, int M, int N, int K, int actGelu)
{
    const int lane = threadIdx.x;                    // 0..31 (wave32)
    const int row0 = blockIdx.y * 128 + threadIdx.y * 16;
    const int col0 = blockIdx.x * 64;
    const int nTiles = N >> 4;
    const int kIters = K >> 5;                       // even for all our shapes
    const int lm = lane & 15;
    const int kb = (lane >> 4) << 4;                 // 0 or 16 (A frag K base)

    v8f acc[4];
    #pragma unroll
    for (int t = 0; t < 4; ++t)
        #pragma unroll
        for (int e = 0; e < 8; ++e) acc[t][e] = 0.0f;

    const __bf16* Arow = A + (size_t)(row0 + lm) * K;
    const __bf16* Bbase = Bp + (((size_t)(col0 >> 4) << 5) + lane) * 16;
    const size_t bKStep = ((size_t)nTiles << 5) * 16;    // elems per k-step
    const size_t bNStep = (size_t)32 * 16;               // elems per n-tile

    union AB { v16bf v; f32x4 f[2]; };
    AB aA[2];
    AB aB[2][4];

    // stage loader: all indices into aA/aB are literal constants after unroll
    #define LOAD_STAGE(st, kt)                                                \
        do {                                                                  \
            const int _k0 = (kt) << 5;                                        \
            aA[st].f[0] = *reinterpret_cast<const f32x4*>(Arow + _k0 + kb);   \
            aA[st].f[1] = *reinterpret_cast<const f32x4*>(Arow + _k0 + kb + 8); \
            const f32x4* _bp = reinterpret_cast<const f32x4*>(                \
                Bbase + (size_t)(kt) * bKStep);                               \
            _Pragma("unroll")                                                 \
            for (int _nt = 0; _nt < 4; ++_nt) {                               \
                const f32x4* _p = reinterpret_cast<const f32x4*>(             \
                    reinterpret_cast<const __bf16*>(_bp) + (size_t)_nt * bNStep); \
                aB[st][_nt].f[0] = _p[0];                                     \
                aB[st][_nt].f[1] = _p[1];                                     \
            }                                                                 \
        } while (0)

    LOAD_STAGE(0, 0);
    for (int kt = 0; kt < kIters; kt += 2) {
        LOAD_STAGE(1, kt + 1);
        #pragma unroll
        for (int nt = 0; nt < 4; ++nt)
            acc[nt] = __builtin_amdgcn_wmma_f32_16x16x32_bf16(
                false, aA[0].v, false, aB[0][nt].v, (short)0, acc[nt], false, false);
        if (kt + 2 < kIters) LOAD_STAGE(0, kt + 2);
        #pragma unroll
        for (int nt = 0; nt < 4; ++nt)
            acc[nt] = __builtin_amdgcn_wmma_f32_16x16x32_bf16(
                false, aA[1].v, false, aB[1][nt].v, (short)0, acc[nt], false, false);
    }
    #undef LOAD_STAGE

    // C/D layout: VGPR r -> row r (lanes 0-15) / row 8+r (lanes 16-31)
    const int mo = (lane >> 4) << 3;
    #pragma unroll
    for (int nt = 0; nt < 4; ++nt) {
        const int col = col0 + nt * 16 + lm;
        const float bv = bias ? bias[col] : 0.0f;
        #pragma unroll
        for (int r = 0; r < 8; ++r) {
            const size_t idx = (size_t)(row0 + mo + r) * N + col;
            float v = acc[nt][r] + bv;
            if (addsrc) v += addsrc[idx];
            if (actGelu) v = gelu_exact(v);
            C[idx] = v;
        }
    }
}

// ---------------------------------------------------------------------------
// Implicit filter MLP: filt_t[c,l] (transposed to (H,L) for contiguous FFT)
// ---------------------------------------------------------------------------
__global__ __launch_bounds__(64) void k_filtmlp(
    const float* __restrict__ fw1, const float* __restrict__ fb1,
    const float* __restrict__ fw2, const float* __restrict__ fb2,
    const float* __restrict__ fw3, const float* __restrict__ fb3,
    float* __restrict__ filt_t)
{
    __shared__ float f1[FHH], f2[FHH];
    const int l = blockIdx.x;
    const int j = threadIdx.x;
    const float t = (float)l * (1.0f / (LL - 1));
    f1[j] = gelu_exact(t * fw1[j] + fb1[j]);
    __syncthreads();
    float s = fb2[j];
    for (int k = 0; k < FHH; ++k) s += f1[k] * fw2[k * FHH + j];
    f2[j] = gelu_exact(s);
    __syncthreads();
    for (int c = j; c < HB; c += FHH) {
        float s3 = fb3[c];
        for (int k = 0; k < FHH; ++k) s3 += f2[k] * fw3[k * HB + c];
        filt_t[(size_t)c * LL + l] = s3;
    }
}

// ---------------------------------------------------------------------------
// Depthwise conv k=3 pad=1 on u (= proj[:, :, H:2H]); writes (B,H,L) layout
// ---------------------------------------------------------------------------
__global__ __launch_bounds__(256) void k_conv(
    const float* __restrict__ proj, const float* __restrict__ cw,
    const float* __restrict__ cb, float* __restrict__ us_t)
{
    int idx = blockIdx.x * 256 + threadIdx.x;  // (b*HB+c)*LL + l
    if (idx >= BB * HB * LL) return;
    int l  = idx & (LL - 1);
    int rc = idx >> 12;
    int c  = rc & (HB - 1);
    int b  = rc >> 9;
    const float* up = proj + ((size_t)b * LL) * H3 + HB + c;
    float um = (l > 0)      ? up[(size_t)(l - 1) * H3] : 0.0f;
    float uc =                up[(size_t)l * H3];
    float ur = (l < LL - 1) ? up[(size_t)(l + 1) * H3] : 0.0f;
    us_t[idx] = um * cw[c * 3 + 0] + uc * cw[c * 3 + 1] + ur * cw[c * 3 + 2] + cb[c];
}

// ---------------------------------------------------------------------------
// In-place radix-2 Cooley-Tukey FFT on 4096 complex points in LDS.
// Input must be in bit-reversed order. sgn=-1: forward, sgn=+1: inverse.
// ---------------------------------------------------------------------------
__device__ __forceinline__ void fft4096_lds(cplx* X, float sgn)
{
    for (int s = 1; s <= 12; ++s) {
        const int mh = 1 << (s - 1);
        const float wstep = sgn * 6.283185307179586f / (float)(1 << s);
        for (int t = threadIdx.x; t < 2048; t += 256) {
            int j = t & (mh - 1);
            int base = ((t >> (s - 1)) << s) + j;
            float sw, cw;
            __sincosf(wstep * (float)j, &sw, &cw);
            cplx u = X[base];
            cplx v = X[base + mh];
            cplx vw; vw.x = v.x * cw - v.y * sw; vw.y = v.x * sw + v.y * cw;
            X[base].x      = u.x + vw.x; X[base].y      = u.y + vw.y;
            X[base + mh].x = u.x - vw.x; X[base + mh].y = u.y - vw.y;
        }
        __syncthreads();
    }
}

// Filter spectrum: per channel c, Fs[c, :] = FFT(filt_t[c, :])
__global__ __launch_bounds__(256) void k_fftfilter(
    const float* __restrict__ filt_t, cplx* __restrict__ Fs)
{
    __shared__ cplx X[LL];
    const int c = blockIdx.x;
    const float* src = filt_t + (size_t)c * LL;
    for (int l = threadIdx.x; l < LL; l += 256) {
        int rv = __brev((unsigned)l) >> 20;
        X[rv].x = src[l]; X[rv].y = 0.0f;
    }
    __syncthreads();
    fft4096_lds(X, -1.0f);
    cplx* dst = Fs + (size_t)c * LL;
    for (int k = threadIdx.x; k < LL; k += 256) dst[k] = X[k];
}

// FFT conv: per (b,c) row: FFT(us) * Fs[c], inverse FFT, real part / N
__global__ __launch_bounds__(256) void k_fftconv(
    const float* __restrict__ us_t, const cplx* __restrict__ Fs,
    float* __restrict__ filtered_t)
{
    __shared__ cplx X[LL];
    const int r = blockIdx.x;            // b*HB + c
    const int c = r & (HB - 1);
    const float* src = us_t + (size_t)r * LL;
    for (int l = threadIdx.x; l < LL; l += 256) {
        int rv = __brev((unsigned)l) >> 20;
        X[rv].x = src[l]; X[rv].y = 0.0f;
    }
    __syncthreads();
    fft4096_lds(X, -1.0f);               // forward
    const cplx* F = Fs + (size_t)c * LL;
    for (int k = threadIdx.x; k < LL; k += 256) {
        cplx a = X[k], f = F[k];
        cplx p; p.x = a.x * f.x - a.y * f.y; p.y = a.x * f.y + a.y * f.x;
        X[k] = p;
    }
    __syncthreads();
    for (int l = threadIdx.x; l < LL; l += 256) {   // re-bit-reverse for inverse
        int rv = __brev((unsigned)l) >> 20;
        if (l < rv) { cplx tmp = X[l]; X[l] = X[rv]; X[rv] = tmp; }
    }
    __syncthreads();
    fft4096_lds(X, 1.0f);                // inverse (scale applied at write)
    float* dst = filtered_t + (size_t)r * LL;
    for (int l = threadIdx.x; l < LL; l += 256)
        dst[l] = X[l].x * (1.0f / (float)LL);
}

// ---------------------------------------------------------------------------
// Gate: out = v * filtered * sigmoid(z)  -> bf16 for the wout GEMM
// ---------------------------------------------------------------------------
__global__ __launch_bounds__(256) void k_gate(
    const float* __restrict__ proj, const float* __restrict__ filtered_t,
    __bf16* __restrict__ gate)
{
    int idx = blockIdx.x * 256 + threadIdx.x;   // over MROWS*HB, c fastest
    if (idx >= MROWS * HB) return;
    int c   = idx & (HB - 1);
    int row = idx >> 9;
    int l   = row & (LL - 1);
    int b   = row >> 12;
    const float* p = proj + (size_t)row * H3;
    float v = p[c];
    float z = p[2 * HB + c];
    float f = filtered_t[((size_t)(b * HB + c)) * LL + l];
    float sig = 1.0f / (1.0f + __expf(-z));
    gate[idx] = (__bf16)(v * f * sig);
}

// ---------------------------------------------------------------------------
// Head: out[row, o] = g1[row] . hw2[:, o] + hb2[o]   (K=256, OUT=2)
// ---------------------------------------------------------------------------
__global__ __launch_bounds__(256) void k_head(
    const float* __restrict__ g1v, const float* __restrict__ hw2,
    const float* __restrict__ hb2, float* __restrict__ out)
{
    int idx = blockIdx.x * 256 + threadIdx.x;   // MROWS*OUTD
    if (idx >= MROWS * OUTD) return;
    int o   = idx & 1;
    int row = idx >> 1;
    const float* gr = g1v + (size_t)row * HHALF;
    float s = hb2[o];
    for (int k = 0; k < HHALF; ++k) s += gr[k] * hw2[k * OUTD + o];
    out[idx] = s;
}

// ---------------------------------------------------------------------------
// Host driver
// ---------------------------------------------------------------------------
extern "C" void kernel_launch(void* const* d_in, const int* in_sizes, int n_in,
                              void* d_out, int out_size, void* d_ws, size_t ws_size,
                              hipStream_t stream)
{
    (void)in_sizes; (void)n_in; (void)out_size; (void)ws_size;
    const float* x        = (const float*)d_in[0];
    const int*   edge_ids = (const int*)  d_in[1];
    const float* in_w     = (const float*)d_in[2];
    const float* in_b     = (const float*)d_in[3];
    const float* edge_emb = (const float*)d_in[4];
    const float* ln_g     = (const float*)d_in[5];
    const float* ln_b     = (const float*)d_in[6];
    const float* fw1      = (const float*)d_in[7];
    const float* fb1      = (const float*)d_in[8];
    const float* fw2      = (const float*)d_in[9];
    const float* fb2      = (const float*)d_in[10];
    const float* fw3      = (const float*)d_in[11];
    const float* fb3      = (const float*)d_in[12];
    const float* cw       = (const float*)d_in[13];
    const float* cb       = (const float*)d_in[14];
    const float* win_w    = (const float*)d_in[15];
    const float* win_b    = (const float*)d_in[16];
    const float* wout_w   = (const float*)d_in[17];
    const float* wout_b   = (const float*)d_in[18];
    const float* fln_g    = (const float*)d_in[19];
    const float* fln_b    = (const float*)d_in[20];
    const float* hw1      = (const float*)d_in[21];
    const float* hb1      = (const float*)d_in[22];
    const float* hw2      = (const float*)d_in[23];
    const float* hb2      = (const float*)d_in[24];
    float* out = (float*)d_out;

    char* ws = (char*)d_ws;
    size_t off = 0;
    auto alloc = [&](size_t bytes) -> void* {
        void* p = ws + off;
        off += (bytes + 255) & ~(size_t)255;
        return p;
    };
    float*  h      = (float*) alloc((size_t)MROWS * HB * 4);
    __bf16* hn     = (__bf16*)alloc((size_t)MROWS * HB * 2);   // LN out (bf16)
    float*  proj   = (float*) alloc((size_t)MROWS * H3 * 4);
    float*  us_t   = (float*) alloc((size_t)BB * HB * LL * 4); // also filtered_t
    float*  filt_t = (float*) alloc((size_t)HB * LL * 4);
    cplx*   Fs     = (cplx*)  alloc((size_t)HB * LL * 8);
    __bf16* win_p  = (__bf16*)alloc((size_t)DEPTH * HB * H3 * 2);
    __bf16* wout_p = (__bf16*)alloc((size_t)DEPTH * HB * HB * 2);
    __bf16* hw1_p  = (__bf16*)alloc((size_t)HB * HHALF * 2);
    // Aliases (lifetimes do not overlap):
    float*  filtered_t = us_t;   // fftconv stages row in LDS before writing
    __bf16* gate = hn;           // hn consumed by GEMM1 before gate is produced
    float*  g1v  = proj;         // proj dead once final LN runs

    // --- pack weights to WMMA fragment layout (bf16) ---
    for (int i = 0; i < DEPTH; ++i) {
        int tw = (HB / 32) * (H3 / 16) * 32;
        k_pack<<<(tw + 255) / 256, 256, 0, stream>>>(
            win_w + (size_t)i * HB * H3, win_p + (size_t)i * HB * H3, HB, H3);
        int tw2 = (HB / 32) * (HB / 16) * 32;
        k_pack<<<(tw2 + 255) / 256, 256, 0, stream>>>(
            wout_w + (size_t)i * HB * HB, wout_p + (size_t)i * HB * HB, HB, HB);
    }
    {
        int tw3 = (HB / 32) * (HHALF / 16) * 32;
        k_pack<<<(tw3 + 255) / 256, 256, 0, stream>>>(hw1, hw1_p, HB, HHALF);
    }

    // --- embed ---
    k_embed<<<(MROWS * HB) / 256, 256, 0, stream>>>(x, edge_ids, in_w, in_b, edge_emb, h);

    const dim3 gblk(32, 8, 1);
    for (int i = 0; i < DEPTH; ++i) {
        k_ln<<<MROWS, 256, 0, stream>>>(h, ln_g + i * HB, ln_b + i * HB, hn);
        dim3 g1g(H3 / 64, MROWS / 128, 1);
        k_gemm<<<g1g, gblk, 0, stream>>>(hn, win_p + (size_t)i * HB * H3,
                                         win_b + i * H3, nullptr, proj,
                                         MROWS, H3, HB, 0);
        k_filtmlp<<<LL, 64, 0, stream>>>(fw1 + i * FHH, fb1 + i * FHH,
                                         fw2 + (size_t)i * FHH * FHH, fb2 + i * FHH,
                                         fw3 + (size_t)i * FHH * HB, fb3 + i * HB,
                                         filt_t);
        k_fftfilter<<<HB, 256, 0, stream>>>(filt_t, Fs);
        k_conv<<<(BB * HB * LL) / 256, 256, 0, stream>>>(
            proj, cw + (size_t)i * HB * 3, cb + i * HB, us_t);
        k_fftconv<<<BB * HB, 256, 0, stream>>>(us_t, Fs, filtered_t);
        k_gate<<<(MROWS * HB) / 256, 256, 0, stream>>>(proj, filtered_t, gate);
        dim3 g2g(HB / 64, MROWS / 128, 1);
        k_gemm<<<g2g, gblk, 0, stream>>>(gate, wout_p + (size_t)i * HB * HB,
                                         wout_b + i * HB, h, h,
                                         MROWS, HB, HB, 0);   // h += out@wout + b
    }

    // --- final LN + MLP head ---
    k_ln<<<MROWS, 256, 0, stream>>>(h, fln_g, fln_b, hn);
    dim3 g3g(HHALF / 64, MROWS / 128, 1);
    k_gemm<<<g3g, gblk, 0, stream>>>(hn, hw1_p, hb1, nullptr, g1v,
                                     MROWS, HHALF, HB, 1);    // gelu fused
    k_head<<<(MROWS * OUTD + 255) / 256, 256, 0, stream>>>(g1v, hw2, hb2, out);
}
